// MinGRUResidualBlock_37452114821246
// MI455X (gfx1250) — matrix operand order repaired and checked by reading, not compiled, not run on hardware
//
#include <hip/hip_runtime.h>
#include <hip/hip_bf16.h>

// ---------------------------------------------------------------------------
// Types for CDNA5 WMMA / TDM (gfx1250, wave32)
// ---------------------------------------------------------------------------
typedef __attribute__((ext_vector_type(16))) __bf16   v16bf;
typedef __attribute__((ext_vector_type(8)))  __bf16   v8bf;
typedef __attribute__((ext_vector_type(8)))  float    v8f;
typedef __attribute__((ext_vector_type(4)))  unsigned v4u;
typedef __attribute__((ext_vector_type(8)))  int      v8i;
typedef __attribute__((ext_vector_type(4)))  int      v4i;

#define DMODEL 1024
#define HDIM   4096
#define BATCH  8
#define SEQ    2048
#define N2     2048   // 2*D

__device__ __forceinline__ __bf16 f2bf(float f) {
    unsigned u = __builtin_bit_cast(unsigned, f);
    unsigned r = (u + 0x7FFFu + ((u >> 16) & 1u)) >> 16;
    unsigned short s = (unsigned short)r;
    return __builtin_bit_cast(__bf16, s);
}

// A-fragment (16x32 bf16, MxK).  ISA layout: lane l (group h = l>>4):
//   row M = l&15 ; V0..V3 hold K = h*8 .. h*8+7 ; V4..V7 hold K = 16+h*8 ..
__device__ __forceinline__ v16bf load_a_frag(const __bf16* base, int ld, int lane) {
    const int r = lane & 15;
    const int h = (lane >> 4) * 8;
    const __bf16* p = base + (size_t)r * ld;
    v8bf lo = *(const v8bf*)(p + h);
    v8bf hi = *(const v8bf*)(p + 16 + h);
    v16bf out;
#pragma unroll
    for (int i = 0; i < 8; ++i) { out[i] = lo[i]; out[i + 8] = hi[i]; }
    return out;
}

// B-fragment (32x16 bf16, KxN) from a B^T ([N][K]) tile.  ISA layout:
//   lane l: col N = l&15 ; V0..V3 hold K = h*8 .. h*8+7 ; V4..V7 hold K = 16+h*8 ..
__device__ __forceinline__ v16bf load_b_frag(const __bf16* baseT, int ld, int lane) {
    const int n  = lane & 15;
    const int kb = (lane >> 4) * 8;
    const __bf16* p = baseT + (size_t)n * ld + kb;
    v8bf lo = *(const v8bf*)(p);
    v8bf hi = *(const v8bf*)(p + 16);
    v16bf out;
#pragma unroll
    for (int i = 0; i < 8; ++i) { out[i] = lo[i]; out[i + 8] = hi[i]; }
    return out;
}

__device__ __forceinline__ v8f wmma_bf16(v16bf a, v16bf b, v8f c) {
    return __builtin_amdgcn_wmma_f32_16x16x32_bf16(
        /*neg_a=*/false, a, /*neg_b=*/false, b,
        /*c_mod=*/(short)0, c, /*reuse_a=*/false, /*reuse_b=*/false);
}

// ---------------------------------------------------------------------------
// Tensor Data Mover: async 2D bf16 tile load  (global -> LDS).
// D# per CDNA5 ISA:
//   group0: count=1 | lds_addr | global_addr(57b) | type=2
//   group1: data_size=2B, tensor_dim0/1, tile_dim0/1, dim0_stride (48b)
// amdgpu-toolchain (clang-23) 6-arg builtin form:
//   (uint32x4 g0, int32x8 g1, int32x4 g2, int32x4 g3, int32x8 g4, i32 cpol)
// ---------------------------------------------------------------------------
__device__ __forceinline__ void tdm_load_tile_2d(
        unsigned lds_addr, const __bf16* gptr,
        unsigned tile_k, unsigned tile_rows,
        unsigned tensor_k, unsigned tensor_rows, unsigned row_stride) {
    unsigned long long ga = (unsigned long long)(size_t)gptr;
    v4u g0;
    g0[0] = 1u;                                           // count=1 (valid), load
    g0[1] = lds_addr;                                     // LDS byte address
    g0[2] = (unsigned)(ga & 0xFFFFFFFFu);                 // global_addr[31:0]
    g0[3] = (unsigned)((ga >> 32) & 0x01FFFFFFu)          // global_addr[56:32]
          | (2u << 30);                                   // type=2 ("image")
    v8i g1;
    g1[0] = (int)(1u << 16);                              // wg_mask=0, data_size=2B
    g1[1] = (int)((tensor_k & 0xFFFFu) << 16);            // tensor_dim0[15:0]
    g1[2] = (int)((tensor_k >> 16) |
                  ((tensor_rows & 0xFFFFu) << 16));       // dim0[31:16] | dim1[15:0]
    g1[3] = (int)((tensor_rows >> 16) |
                  ((tile_k & 0xFFFFu) << 16));            // dim1[31:16] | tile_dim0
    g1[4] = (int)(tile_rows & 0xFFFFu);                   // tile_dim1 (tile_dim2=0)
    g1[5] = (int)row_stride;                              // tensor_dim0_stride[31:0]
    g1[6] = 0;                                            // stride hi / dim1_stride lo
    g1[7] = 0;                                            // dim1_stride hi
    v4i z4 = {0, 0, 0, 0};
    v8i z8 = {0, 0, 0, 0, 0, 0, 0, 0};
    __builtin_amdgcn_tensor_load_to_lds(g0, g1, z4, z4, z8, 0);
}

// ---------------------------------------------------------------------------
// fp32 -> bf16 weight conversion (weights are [N][K] row-major = B^T already)
// ---------------------------------------------------------------------------
__global__ __launch_bounds__(256) void convert_bf16_kernel(
        const float* __restrict__ src, __bf16* __restrict__ dst, int n) {
    int i = blockIdx.x * 256 + threadIdx.x;
    if (i < n) dst[i] = f2bf(src[i]);
}

// ---------------------------------------------------------------------------
// LayerNorm over rows of [R,1024] -> bf16
// ---------------------------------------------------------------------------
__global__ __launch_bounds__(256) void ln_bf16_kernel(
        const float* __restrict__ x, const float* __restrict__ w,
        const float* __restrict__ b, __bf16* __restrict__ out) {
    __shared__ float red[256];
    const int tid = threadIdx.x;
    const float* xr = x + (size_t)blockIdx.x * DMODEL;

    float s = 0.f;
    for (int i = tid; i < DMODEL; i += 256) s += xr[i];
    red[tid] = s; __syncthreads();
    for (int o = 128; o > 0; o >>= 1) {
        if (tid < o) red[tid] += red[tid + o];
        __syncthreads();
    }
    const float mu = red[0] * (1.f / DMODEL);
    __syncthreads();

    float v = 0.f;
    for (int i = tid; i < DMODEL; i += 256) { float d = xr[i] - mu; v += d * d; }
    red[tid] = v; __syncthreads();
    for (int o = 128; o > 0; o >>= 1) {
        if (tid < o) red[tid] += red[tid + o];
        __syncthreads();
    }
    const float rs = rsqrtf(red[0] * (1.f / DMODEL) + 1e-5f);

    __bf16* orow = out + (size_t)blockIdx.x * DMODEL;
    for (int i = tid; i < DMODEL; i += 256)
        orow[i] = f2bf((xr[i] - mu) * rs * w[i] + b[i]);
}

// ---------------------------------------------------------------------------
// Init recurrence state: hf = state (fp32), hbf0/hbf1 = bf16(state) with
// rows 8..15 zero-padded (WMMA M=16 tile, only 8 batches valid).
// ---------------------------------------------------------------------------
__global__ __launch_bounds__(256) void init_h_kernel(
        const float* __restrict__ state, float* __restrict__ hf,
        __bf16* __restrict__ hbf0, __bf16* __restrict__ hbf1) {
    int i = blockIdx.x * 256 + threadIdx.x;     // 16*1024 threads
    int row = i >> 10;
    if (row < BATCH) {
        float v = state[i];
        hf[i] = v;
        __bf16 bv = f2bf(v);
        hbf0[i] = bv; hbf1[i] = bv;
    } else {
        __bf16 z = f2bf(0.f);
        hbf0[i] = z; hbf1[i] = z;
    }
}

// ---------------------------------------------------------------------------
// Tiled bf16 WMMA GEMM with TDM double-buffered LDS staging:
//   C[M,N] = A[M,K] @ W[N,K]^T + bias
// Tile: 128x128x32, 256 threads = 8 waves, each wave a 32x64 region
//       (2x4 grid of 16x16 WMMA tiles, 32 K per WMMA).
// Wave 0 drives the Tensor Data Mover: two 2D tile descriptors per K-step
// (A tile 128x32, W^T tile 128x32), double-buffered; steady state waits only
// TENSORcnt<=2 so the next buffer's DMA overlaps the current buffer's WMMAs.
//   EPI 0: C fp32                    (in_proj "pre")
//   EPI 1: C = bf16(gelu_exact(C))   (MLP hidden)
//   EPI 2: C fp32 = C + resid        (final residual -> d_out)
// ---------------------------------------------------------------------------
template <int EPI>
__global__ __launch_bounds__(256) void gemm_bf16_kernel(
        const __bf16* __restrict__ A, const __bf16* __restrict__ W,
        const float* __restrict__ bias, const float* __restrict__ resid,
        float* __restrict__ Cf, __bf16* __restrict__ Cb,
        int M, int N, int K) {
    const int tid  = threadIdx.x;
    const int lane = tid & 31;
    const int wave = tid >> 5;
    const size_t bm = (size_t)blockIdx.x * 128;
    const size_t bn = (size_t)blockIdx.y * 128;

    __shared__ __align__(32) __bf16 sA[2][128 * 32];   // [m][k] double-buffered
    __shared__ __align__(32) __bf16 sB[2][128 * 32];   // [n][k] (B^T tile)

    const int wrow = (wave & 3) * 32;
    const int wcol = (wave >> 2) * 64;

    v8f acc[2][4];
#pragma unroll
    for (int a = 0; a < 2; ++a)
#pragma unroll
        for (int b = 0; b < 4; ++b) acc[a][b] = (v8f){0.f,0.f,0.f,0.f,0.f,0.f,0.f,0.f};

    const __bf16* Abase = A + bm * K;
    const __bf16* Wbase = W + bn * K;
    const unsigned ldsA0 = (unsigned)(size_t)(void*)&sA[0][0];
    const unsigned ldsA1 = (unsigned)(size_t)(void*)&sA[1][0];
    const unsigned ldsB0 = (unsigned)(size_t)(void*)&sB[0][0];
    const unsigned ldsB1 = (unsigned)(size_t)(void*)&sB[1][0];

    // Prologue: DMA first A/W tiles into buffer 0.
    if (wave == 0) {
        tdm_load_tile_2d(ldsA0, Abase, 32, 128, (unsigned)K, 128, (unsigned)K);
        tdm_load_tile_2d(ldsB0, Wbase, 32, 128, (unsigned)K, 128, (unsigned)K);
    }

    for (int k0 = 0; k0 < K; k0 += 32) {
        const int cur = (k0 >> 5) & 1;
        if (wave == 0) {
            if (k0 + 32 < K) {
                // Kick off next buffer's tiles, then wait only for current pair.
                const unsigned la = cur ? ldsA0 : ldsA1;
                const unsigned lb = cur ? ldsB0 : ldsB1;
                tdm_load_tile_2d(la, Abase + k0 + 32, 32, 128, (unsigned)K, 128, (unsigned)K);
                tdm_load_tile_2d(lb, Wbase + k0 + 32, 32, 128, (unsigned)K, 128, (unsigned)K);
                __builtin_amdgcn_s_wait_tensorcnt(2);
            } else {
                __builtin_amdgcn_s_wait_tensorcnt(0);
            }
        }
        __syncthreads();

        const __bf16* cA = sA[cur];
        const __bf16* cB = sB[cur];
        v16bf af[2], bf_[4];
#pragma unroll
        for (int a = 0; a < 2; ++a)
            af[a] = load_a_frag(cA + (wrow + a * 16) * 32, 32, lane);
#pragma unroll
        for (int b = 0; b < 4; ++b)
            bf_[b] = load_b_frag(cB + (wcol + b * 16) * 32, 32, lane);
#pragma unroll
        for (int a = 0; a < 2; ++a)
#pragma unroll
            for (int b = 0; b < 4; ++b)
                acc[a][b] = wmma_bf16(af[a], bf_[b], acc[a][b]);
        __syncthreads();
    }

    // Epilogue. C tile layout: VGPR i, lane l -> row i + (l>>4)*8, col l&15.
    const int rsub = (lane >> 4) * 8;
    const int csub = lane & 15;
#pragma unroll
    for (int a = 0; a < 2; ++a) {
#pragma unroll
        for (int b = 0; b < 4; ++b) {
            const size_t col = bn + wcol + b * 16 + csub;
            const float bv = bias[col];
#pragma unroll
            for (int i = 0; i < 8; ++i) {
                const size_t row = bm + wrow + a * 16 + rsub + i;
                float v = acc[a][b][i] + bv;
                if (EPI == 0) {
                    Cf[row * N + col] = v;
                } else if (EPI == 1) {
                    float g = 0.5f * v * (1.f + erff(v * 0.70710678118654752f));
                    Cb[row * N + col] = f2bf(g);
                } else {
                    Cf[row * N + col] = v + resid[row * N + col];
                }
            }
        }
    }
}

// ---------------------------------------------------------------------------
// Device-scope sense barrier for the persistent recurrence kernel.
// ---------------------------------------------------------------------------
__device__ __forceinline__ void grid_barrier(unsigned* cnt, unsigned* gen, unsigned nwg) {
    __syncthreads();
    if (threadIdx.x == 0) {
        unsigned g = __hip_atomic_load(gen, __ATOMIC_ACQUIRE, __HIP_MEMORY_SCOPE_AGENT);
        __threadfence();
        unsigned arrived = __hip_atomic_fetch_add(cnt, 1u, __ATOMIC_ACQ_REL,
                                                  __HIP_MEMORY_SCOPE_AGENT);
        if (arrived == nwg - 1) {
            __hip_atomic_store(cnt, 0u, __ATOMIC_RELAXED, __HIP_MEMORY_SCOPE_AGENT);
            __hip_atomic_fetch_add(gen, 1u, __ATOMIC_ACQ_REL, __HIP_MEMORY_SCOPE_AGENT);
        } else {
            while (__hip_atomic_load(gen, __ATOMIC_ACQUIRE,
                                     __HIP_MEMORY_SCOPE_AGENT) == g)
                __builtin_amdgcn_s_sleep(2);
        }
    }
    __syncthreads();
}

// ---------------------------------------------------------------------------
// Persistent minGRU recurrence.  64 WGs x 64 threads (2 waves).
// WG g owns gate columns [g*16, g*16+16): wave0 -> u half (Ws rows j),
// wave1 -> c half (Ws rows D+j).  Its 32x1024 slice of Ws lives in LDS as
// bf16 (64 KB of 320 KB/WGP).  Each step: two 16x16x1024 WMMA GEMMs against
// the (double-buffered) bf16 h state, then sigmoid/tanh/lerp, write rec and
// new h, grid barrier.
// ---------------------------------------------------------------------------
__global__ __launch_bounds__(64) void recurrence_kernel(
        const float* __restrict__ pre,      // [B,S,2D] fp32
        const float* __restrict__ Ws,       // [2D,D]   fp32
        float* __restrict__ hf,             // [B,D]    fp32 running state
        __bf16* __restrict__ hbf0,          // [16,D]   bf16 state buf 0
        __bf16* __restrict__ hbf1,          // [16,D]   bf16 state buf 1
        float* __restrict__ rec,            // [B,S,D]  fp32 scan outputs
        float* __restrict__ hlast,          // [B,D]    fp32 -> d_out tail
        unsigned* __restrict__ ctrl) {
    const int g    = blockIdx.x;            // 0..63
    const int tid  = threadIdx.x;           // 0..63
    const int lane = tid & 31;
    const int wave = tid >> 5;              // 0 = u, 1 = c
    const int j0   = g * 16;

    __shared__ __align__(32) __bf16 sW[32 * DMODEL];   // 64 KB: 16 u rows + 16 c rows
    __shared__ float sU[BATCH * 16];
    __shared__ float sC[BATCH * 16];

    // Load + convert the Ws slice once (B^T layout: [row][k], k contiguous).
    for (int idx = tid; idx < 32 * DMODEL; idx += 64) {
        int r = idx >> 10, k = idx & (DMODEL - 1);
        int src = (r < 16) ? (j0 + r) : (DMODEL + j0 + (r - 16));
        sW[idx] = f2bf(Ws[(size_t)src * DMODEL + k]);
    }
    __syncthreads();

    const __bf16* myW = sW + (wave ? 16 * DMODEL : 0);
    unsigned* cnt = ctrl;
    unsigned* gen = ctrl + 1;

    for (int s = 0; s < SEQ; ++s) {
        const __bf16* hread  = (s & 1) ? hbf1 : hbf0;
        __bf16*       hwrite = (s & 1) ? hbf0 : hbf1;

        // 16x16 tile (M=16 padded batch) x K=1024: 32 WMMAs per wave.
        v8f acc = (v8f){0.f,0.f,0.f,0.f,0.f,0.f,0.f,0.f};
#pragma unroll 4
        for (int k0 = 0; k0 < DMODEL; k0 += 32) {
            v16bf a  = load_a_frag(hread + k0, DMODEL, lane);
            v16bf bb = load_b_frag(myW + k0, DMODEL, lane);
            acc = wmma_bf16(a, bb, acc);
        }

        // Spill this wave's tile (rows 0..7 valid) so the two waves can mix.
        float* dst = wave ? sC : sU;
        const int rsub = (lane >> 4) * 8;
        const int csub = lane & 15;
#pragma unroll
        for (int i = 0; i < 8; ++i) {
            int r = rsub + i;
            if (r < BATCH) dst[r * 16 + csub] = acc[i];
        }
        __syncthreads();

        // Elementwise minGRU update for this WG's 8x16 slice.
        for (int e = tid; e < BATCH * 16; e += 64) {
            int b = e >> 4, jj = e & 15, j = j0 + jj;
            size_t pb = ((size_t)b * SEQ + s) * N2;
            float du = sU[e] + pre[pb + j];
            float dc = sC[e] + pre[pb + DMODEL + j];
            float u = 1.f / (1.f + __expf(-du));
            float c = tanhf(dc);
            float h = hf[b * DMODEL + j];
            float hn = h + u * (c - h);
            hf[b * DMODEL + j] = hn;
            hwrite[b * DMODEL + j] = f2bf(hn);
            rec[((size_t)b * SEQ + s) * DMODEL + j] = hn;
            if (s == SEQ - 1) hlast[b * DMODEL + j] = hn;
        }

        grid_barrier(cnt, gen, gridDim.x);
    }
}

// ---------------------------------------------------------------------------
// Launch
// ---------------------------------------------------------------------------
extern "C" void kernel_launch(void* const* d_in, const int* in_sizes, int n_in,
                              void* d_out, int out_size, void* d_ws, size_t ws_size,
                              hipStream_t stream) {
    (void)in_sizes; (void)n_in; (void)out_size; (void)ws_size;
    const float* x     = (const float*)d_in[0];
    const float* state = (const float*)d_in[1];
    const float* ln1w  = (const float*)d_in[2];
    const float* ln1b  = (const float*)d_in[3];
    const float* Win   = (const float*)d_in[4];
    const float* bin_  = (const float*)d_in[5];
    const float* Ws    = (const float*)d_in[6];
    const float* ln2w  = (const float*)d_in[7];
    const float* ln2b  = (const float*)d_in[8];
    const float* W1    = (const float*)d_in[9];
    const float* b1    = (const float*)d_in[10];
    const float* W2    = (const float*)d_in[11];
    const float* b2    = (const float*)d_in[12];

    const size_t M = (size_t)BATCH * SEQ;     // 16384 rows
    char* ws = (char*)d_ws;
    size_t off = 0;
    auto alloc = [&](size_t bytes) {
        void* p = ws + off;
        off = (off + bytes + 255) & ~(size_t)255;
        return p;
    };
    unsigned* ctrl   = (unsigned*)alloc(256);
    float*    hf     = (float*)  alloc((size_t)BATCH * DMODEL * 4);
    __bf16*   hbf0   = (__bf16*) alloc((size_t)16 * DMODEL * 2);
    __bf16*   hbf1   = (__bf16*) alloc((size_t)16 * DMODEL * 2);
    __bf16*   win_bf = (__bf16*) alloc((size_t)N2 * DMODEL * 2);
    __bf16*   w1_bf  = (__bf16*) alloc((size_t)HDIM * DMODEL * 2);
    __bf16*   w2_bf  = (__bf16*) alloc((size_t)DMODEL * HDIM * 2);
    __bf16*   xn     = (__bf16*) alloc(M * DMODEL * 2);
    __bf16*   yn     = (__bf16*) alloc(M * DMODEL * 2);
    float*    pre    = (float*)  alloc(M * N2 * 4);
    float*    rec    = (float*)  alloc(M * DMODEL * 4);
    __bf16*   act    = (__bf16*) alloc(M * HDIM * 2);

    float* out   = (float*)d_out;
    float* hlast = out + M * DMODEL;          // tuple tail: h_last [B,D]

    (void)hipMemsetAsync(ctrl, 0, 256, stream);

    // Weight conversions (weights already [N][K] == B^T layout).
    {
        int n = N2 * DMODEL;
        convert_bf16_kernel<<<(n + 255) / 256, 256, 0, stream>>>(Win, win_bf, n);
        n = HDIM * DMODEL;
        convert_bf16_kernel<<<(n + 255) / 256, 256, 0, stream>>>(W1, w1_bf, n);
        n = DMODEL * HDIM;
        convert_bf16_kernel<<<(n + 255) / 256, 256, 0, stream>>>(W2, w2_bf, n);
    }
    init_h_kernel<<<(16 * DMODEL) / 256, 256, 0, stream>>>(state, hf, hbf0, hbf1);

    // LN1 -> xn (bf16)
    ln_bf16_kernel<<<(unsigned)M, 256, 0, stream>>>(x, ln1w, ln1b, xn);

    // pre = xn @ Win^T + bin   (fp32)
    gemm_bf16_kernel<0><<<dim3(M / 128, N2 / 128), 256, 0, stream>>>(
        xn, win_bf, bin_, nullptr, pre, nullptr, (int)M, N2, DMODEL);

    // Sequential minGRU scan (persistent, 64 WGs).
    recurrence_kernel<<<64, 64, 0, stream>>>(pre, Ws, hf, hbf0, hbf1, rec, hlast, ctrl);

    // LN2 -> yn (bf16)
    ln_bf16_kernel<<<(unsigned)M, 256, 0, stream>>>(rec, ln2w, ln2b, yn);

    // act = gelu(yn @ W1^T + b1)   (bf16)
    gemm_bf16_kernel<1><<<dim3(M / 128, HDIM / 128), 256, 0, stream>>>(
        yn, w1_bf, b1, nullptr, nullptr, act, (int)M, HDIM, DMODEL);

    // out = x + act @ W2^T + b2    (fp32)
    gemm_bf16_kernel<2><<<dim3(M / 128, DMODEL / 128), 256, 0, stream>>>(
        act, w2_bf, b2, x, out, nullptr, (int)M, DMODEL, HDIM);
}
